// TimeMix_13030930776722
// MI455X (gfx1250) — compile-verified, matrix operand order checked by reading, not run on hardware
//
#include <hip/hip_runtime.h>
#include <hip/hip_bf16.h>
#include <stdint.h>

// ---------------------------------------------------------------------------
// RWKV TimeMix on MI455X (gfx1250, wave32, WMMA).
// B=8, T=2048, C=1024.
//  - GEMMs: v_wmma_f32_16x16x32_bf16, LDS double-buffered tiles filled with
//    GLOBAL_LOAD_ASYNC_TO_LDS_B128 (ASYNCcnt), ds_load_b128 fragment reads.
//    Block tile 128x256, wave tile 64x64 (16 accumulators) -> 1:1 ds:wmma.
//  - WKV: chunk-parallel associative scan (16 chunks of 128), log-space stable.
// ---------------------------------------------------------------------------

typedef __attribute__((ext_vector_type(16))) __bf16 v16bf;
typedef __attribute__((ext_vector_type(8)))  __bf16 v8bf;
typedef __attribute__((ext_vector_type(8)))  float  v8f;

#define BB 8
#define TT 2048
#define CC 1024
#define MM (BB * TT)            // 16384 rows
#define NCH 16                  // WKV chunks
#define CLEN (TT / NCH)         // 128 steps per chunk
#define NEGF (-1e30f)
#define LDSPAD 40               // row stride (elems): 80B -> conflict-free banks

// float -> bf16, round to nearest even (NaN/Inf truncated)
__device__ __forceinline__ unsigned short f2bf(float f) {
  union { float f; uint32_t u; } cv; cv.f = f;
  uint32_t u = cv.u;
  uint32_t r = u + 0x7FFFu + ((u >> 16) & 1u);
  if ((u & 0x7F800000u) == 0x7F800000u) r = u;
  return (unsigned short)(r >> 16);
}

// ---------------------------------------------------------------------------
// 1) time-shift + lerp mix, emit bf16 xk/xv/xr
// ---------------------------------------------------------------------------
__global__ __launch_bounds__(256) void prep_mix_kernel(
    const float* __restrict__ x,
    const float* __restrict__ mk, const float* __restrict__ mv,
    const float* __restrict__ mr,
    unsigned short* __restrict__ xk, unsigned short* __restrict__ xv,
    unsigned short* __restrict__ xr) {
  size_t idx = (size_t)blockIdx.x * blockDim.x + threadIdx.x;  // over B*T*C
  int c = (int)(idx & (CC - 1));
  size_t bt = idx >> 10;
  int t = (int)(bt & (TT - 1));
  float xc = x[idx];
  float xp = (t == 0) ? 0.f : x[idx - CC];
  float d = xc - xp;
  xk[idx] = f2bf(xp + mk[c] * d);
  xv[idx] = f2bf(xp + mv[c] * d);
  xr[idx] = f2bf(xp + mr[c] * d);
}

// ---------------------------------------------------------------------------
// 2) weight fp32 -> bf16
// ---------------------------------------------------------------------------
__global__ __launch_bounds__(256) void cvt_w_kernel(
    const float* __restrict__ a, const float* __restrict__ b,
    const float* __restrict__ c, const float* __restrict__ d,
    unsigned short* __restrict__ oa, unsigned short* __restrict__ ob,
    unsigned short* __restrict__ oc, unsigned short* __restrict__ od) {
  size_t idx = (size_t)blockIdx.x * blockDim.x + threadIdx.x;  // over C*C
  oa[idx] = f2bf(a[idx]);
  ob[idx] = f2bf(b[idx]);
  oc[idx] = f2bf(c[idx]);
  od[idx] = f2bf(d[idx]);
}

// ---------------------------------------------------------------------------
// 3) bf16 WMMA GEMM:  out[m,d] = sum_c A[m,c] * W[d,c]   (both K-contiguous)
//    block tile 128x256 (8 waves, 2x4), wave tile 64x64, K stepped by 32.
//    A/B tiles staged in LDS (double buffer) via async copies.
// ---------------------------------------------------------------------------
__device__ __forceinline__ void async_cp16(const __bf16* gbase, uint32_t goff,
                                           const __bf16* ldsdst) {
  uint32_t loff = (uint32_t)(uintptr_t)ldsdst;   // low 32 bits = LDS offset
  asm volatile("global_load_async_to_lds_b128 %0, %1, %2"
               :: "v"(loff), "v"(goff), "s"(gbase) : "memory");
}

template <int EPI>   // 0 = plain, 1 = sigmoid
__global__ __launch_bounds__(256) void gemm_bf16_kernel(
    const __bf16* __restrict__ A, const __bf16* __restrict__ W,
    float* __restrict__ out) {
  __shared__ __align__(16) __bf16 Abuf[2][128][LDSPAD];
  __shared__ __align__(16) __bf16 Bbuf[2][256][LDSPAD];

  const int tid = threadIdx.x;
  const int lane = tid & 31;
  const int wave = tid >> 5;
  const int l16 = lane & 15, half = lane >> 4;
  const int wm = wave & 1, wn = wave >> 1;           // 2 x 4 wave grid
  const int blockM = blockIdx.x * 128;
  const int blockN = blockIdx.y * 256;

  // A tile: 128 rows x 32 K = 512 x 16B chunks (2/thread, rows r0 and r0+64)
  // B tile: 256 rows x 32 K = 1024 x 16B chunks (4/thread)
  const int r0 = tid >> 2;            // rows 0..63
  const int cc0 = (tid & 3) * 8;      // K element offset within 32

  auto issue = [&](int buf, int kk) {
    uint32_t ga = (uint32_t)(((blockM + r0) * CC + kk + cc0) * 2);
    async_cp16(A, ga, &Abuf[buf][r0][cc0]);
    async_cp16(A, ga + 64u * CC * 2u, &Abuf[buf][r0 + 64][cc0]);
    uint32_t gb = (uint32_t)(((blockN + r0) * CC + kk + cc0) * 2);
#pragma unroll
    for (int s = 0; s < 4; ++s)
      async_cp16(W, gb + (uint32_t)s * 64u * CC * 2u, &Bbuf[buf][r0 + s * 64][cc0]);
  };

  v8f acc[4][4];
  v8f zero = {};
#pragma unroll
  for (int i = 0; i < 4; ++i)
#pragma unroll
    for (int j = 0; j < 4; ++j) acc[i][j] = zero;

  issue(0, 0);
  asm volatile("s_wait_asynccnt 0x0" ::: "memory");
  __syncthreads();

  for (int kk = 0; kk < CC; kk += 32) {
    const int cur = (kk >> 5) & 1;
    if (kk + 32 < CC) issue(cur ^ 1, kk + 32);

    // ISA 16-bit A/B fragment layout: lane<16 -> row=lane, K = {0..7,16..23};
    // lane>=16 -> same rows, K = {8..15,24..31}; i.e. chunks at half*8, +16.
    v16bf a[4], bfr[4];
#pragma unroll
    for (int i = 0; i < 4; ++i) {
      const __bf16* p = &Abuf[cur][wm * 64 + i * 16 + l16][half * 8];
      v8bf u0 = *(const v8bf*)p;
      v8bf u1 = *(const v8bf*)(p + 16);
#pragma unroll
      for (int e = 0; e < 8; ++e) { a[i][e] = u0[e]; a[i][e + 8] = u1[e]; }
    }
#pragma unroll
    for (int j = 0; j < 4; ++j) {
      const __bf16* p = &Bbuf[cur][wn * 64 + j * 16 + l16][half * 8];
      v8bf u0 = *(const v8bf*)p;
      v8bf u1 = *(const v8bf*)(p + 16);
#pragma unroll
      for (int e = 0; e < 8; ++e) { bfr[j][e] = u0[e]; bfr[j][e + 8] = u1[e]; }
    }
#pragma unroll
    for (int i = 0; i < 4; ++i)
#pragma unroll
      for (int j = 0; j < 4; ++j)
        acc[i][j] = __builtin_amdgcn_wmma_f32_16x16x32_bf16(
            false, a[i], false, bfr[j], (short)0, acc[i][j], false, false);

    asm volatile("s_wait_asynccnt 0x0" ::: "memory");
    __syncthreads();
  }

  const int m0 = blockM + wm * 64;
  const int n0 = blockN + wn * 64;
#pragma unroll
  for (int i = 0; i < 4; ++i) {
    int mrow = m0 + i * 16 + half * 8;
#pragma unroll
    for (int j = 0; j < 4; ++j) {
      int ncol = n0 + j * 16 + l16;
#pragma unroll
      for (int r = 0; r < 8; ++r) {
        float vv = acc[i][j][r];
        if (EPI == 1) vv = 1.0f / (1.0f + expf(-vv));
        out[(size_t)(mrow + r) * CC + ncol] = vv;
      }
    }
  }
}

// ---------------------------------------------------------------------------
// 4) WKV chunk-parallel scan.  State (p,q,o) represents num=p*e^o, den=q*e^o.
//    Per-step decay is constant lambda=e^{w_exp} per channel, so a chunk is
//    an affine map:  S_out = Decay_CLEN(S_in) (+) S_chunk, where Decay just
//    adds CLEN*w_exp to o (stays in log space -> no underflow).
// ---------------------------------------------------------------------------

// pass 1: per (chunk, b, c) compute chunk summary from identity state
__global__ __launch_bounds__(256) void wkv_pass1_kernel(
    const float* __restrict__ w,
    const float* __restrict__ k, const float* __restrict__ v,
    float* __restrict__ ps, float* __restrict__ qs, float* __restrict__ os) {
  int idx = blockIdx.x * blockDim.x + threadIdx.x;   // [0, NCH*B*C)
  int c = idx & (CC - 1);
  int bj = idx >> 10;
  int b = bj & (BB - 1);
  int j = bj >> 3;
  float w_exp = -expf(w[c]);
  float p = 0.f, q = 0.f, o = NEGF;
  size_t base = (size_t)b * TT * CC + (size_t)j * CLEN * CC + c;
  for (int t = 0; t < CLEN; ++t) {
    size_t off = base + (size_t)t * CC;
    float kt = k[off], vt = v[off];
    float wo = w_exp + o;
    float no = fmaxf(wo, kt);
    float A2 = expf(wo - no), B2 = expf(kt - no);
    p = A2 * p + B2 * vt;
    q = A2 * q + B2;
    o = no;
  }
  ps[idx] = p; qs[idx] = q; os[idx] = o;
}

// pass 2: per (b, c) exclusive scan of chunk summaries (in place:
// summary[j] is read, then overwritten with the incoming state for chunk j)
__global__ __launch_bounds__(256) void wkv_pass2_kernel(
    const float* __restrict__ w,
    float* __restrict__ ps, float* __restrict__ qs, float* __restrict__ os) {
  int idx = blockIdx.x * blockDim.x + threadIdx.x;   // [0, B*C)
  int c = idx & (CC - 1);
  float Lw = (float)CLEN * (-expf(w[c]));
  float p = 0.f, q = 0.f, o = NEGF;
  for (int j = 0; j < NCH; ++j) {
    int sid = j * (BB * CC) + idx;
    float pc = ps[sid], qc = qs[sid], oc = os[sid];  // chunk summary
    ps[sid] = p; qs[sid] = q; os[sid] = o;           // incoming state
    float od = o + Lw;
    float no = fmaxf(od, oc);
    float ea = expf(od - no), eb = expf(oc - no);
    p = ea * p + eb * pc;
    q = ea * q + eb * qc;
    o = no;
  }
}

// pass 3: replay each chunk from its incoming state, emit gated bf16 output
__global__ __launch_bounds__(256) void wkv_pass3_kernel(
    const float* __restrict__ w, const float* __restrict__ u,
    const float* __restrict__ k, const float* __restrict__ v,
    const float* __restrict__ r,
    const float* __restrict__ ps, const float* __restrict__ qs,
    const float* __restrict__ os, unsigned short* __restrict__ rwkv) {
  int idx = blockIdx.x * blockDim.x + threadIdx.x;   // [0, NCH*B*C)
  int c = idx & (CC - 1);
  int bj = idx >> 10;
  int b = bj & (BB - 1);
  int j = bj >> 3;
  float w_exp = -expf(w[c]);
  float uc = u[c];
  float p = ps[idx], q = qs[idx], o = os[idx];
  size_t base = (size_t)b * TT * CC + (size_t)j * CLEN * CC + c;
  for (int t = 0; t < CLEN; ++t) {
    size_t off = base + (size_t)t * CC;
    float kt = k[off], vt = v[off];
    float uk = uc + kt;
    float no = fmaxf(o, uk);
    float Aa = expf(o - no), Bc = expf(uk - no);
    float y = (Aa * p + Bc * vt) / (Aa * q + Bc);
    rwkv[off] = f2bf(r[off] * y);
    float wo = w_exp + o;
    float no2 = fmaxf(wo, kt);
    float A2 = expf(wo - no2), B2 = expf(kt - no2);
    p = A2 * p + B2 * vt;
    q = A2 * q + B2;
    o = no2;
  }
}

// ---------------------------------------------------------------------------
// launcher
// ---------------------------------------------------------------------------
extern "C" void kernel_launch(void* const* d_in, const int* in_sizes, int n_in,
                              void* d_out, int out_size, void* d_ws, size_t ws_size,
                              hipStream_t stream) {
  const float* x  = (const float*)d_in[0];
  const float* w  = (const float*)d_in[1];
  const float* u  = (const float*)d_in[2];
  const float* mk = (const float*)d_in[3];
  const float* mv = (const float*)d_in[4];
  const float* mr = (const float*)d_in[5];
  const float* Wk = (const float*)d_in[6];
  const float* Wv = (const float*)d_in[7];
  const float* Wr = (const float*)d_in[8];
  const float* Wo = (const float*)d_in[9];
  float* out = (float*)d_out;

  char* ws = (char*)d_ws;
  size_t off = 0;
  const size_t MTC = (size_t)MM * CC;
  const size_t WSZ = (size_t)CC * CC;
  const size_t SSZ = (size_t)NCH * BB * CC;
  unsigned short* xk  = (unsigned short*)(ws + off); off += MTC * 2;
  unsigned short* xv  = (unsigned short*)(ws + off); off += MTC * 2;
  unsigned short* xr  = (unsigned short*)(ws + off); off += MTC * 2;
  unsigned short* wkb = (unsigned short*)(ws + off); off += WSZ * 2;
  unsigned short* wvb = (unsigned short*)(ws + off); off += WSZ * 2;
  unsigned short* wrb = (unsigned short*)(ws + off); off += WSZ * 2;
  unsigned short* wob = (unsigned short*)(ws + off); off += WSZ * 2;
  float* kf = (float*)(ws + off); off += MTC * 4;
  float* vf = (float*)(ws + off); off += MTC * 4;
  float* rf = (float*)(ws + off); off += MTC * 4;
  float* ps = (float*)(ws + off); off += SSZ * 4;
  float* qs = (float*)(ws + off); off += SSZ * 4;
  float* os = (float*)(ws + off); off += SSZ * 4;
  unsigned short* rwkvb = xk;  // xk is dead after the k-GEMM; reuse its buffer

  prep_mix_kernel<<<(unsigned)(MTC / 256), 256, 0, stream>>>(x, mk, mv, mr, xk, xv, xr);
  cvt_w_kernel<<<(unsigned)(WSZ / 256), 256, 0, stream>>>(Wk, Wv, Wr, Wo, wkb, wvb, wrb, wob);

  dim3 gg(MM / 128, CC / 256);
  gemm_bf16_kernel<0><<<gg, 256, 0, stream>>>((const __bf16*)xk, (const __bf16*)wkb, kf);
  gemm_bf16_kernel<0><<<gg, 256, 0, stream>>>((const __bf16*)xv, (const __bf16*)wvb, vf);
  gemm_bf16_kernel<1><<<gg, 256, 0, stream>>>((const __bf16*)xr, (const __bf16*)wrb, rf);

  wkv_pass1_kernel<<<(unsigned)(SSZ / 256), 256, 0, stream>>>(w, kf, vf, ps, qs, os);
  wkv_pass2_kernel<<<(BB * CC) / 256, 256, 0, stream>>>(w, ps, qs, os);
  wkv_pass3_kernel<<<(unsigned)(SSZ / 256), 256, 0, stream>>>(w, u, kf, vf, rf, ps, qs, os, rwkvb);

  gemm_bf16_kernel<0><<<gg, 256, 0, stream>>>((const __bf16*)rwkvb, (const __bf16*)wob, out);
}
